// Attention_9363028705449
// MI455X (gfx1250) — matrix operand order repaired and checked by reading, not compiled, not run on hardware
//
#include <hip/hip_runtime.h>

// ---------------------------------------------------------------------------
// CDNA5 / gfx1250 wave32 WMMA attention pipeline
//   k1: qkv   = bf16( x @ Wqkv ), q-cols pre-scaled by 1/8
//   k2: attn  = bf16( softmax(q k^T) v )  per (b,h,f), n=196 padded to 224
//   k3: out   = f32 ( attn @ Wout + bout )
// ---------------------------------------------------------------------------

typedef __attribute__((ext_vector_type(16))) __bf16 v16bf;
typedef __attribute__((ext_vector_type(8)))  float  v8f;
typedef int v4i __attribute__((vector_size(16)));   // matches builtin param type

union BFrag { v16bf v; uint4 q[2]; unsigned short h[16]; };

__device__ __forceinline__ unsigned short f2bf(float f) {
  unsigned int u = __float_as_uint(f);
  u += 0x7fffu + ((u >> 16) & 1u);          // round-to-nearest-even
  return (unsigned short)(u >> 16);
}

__device__ __forceinline__ v8f wmma_bf16(v16bf a, v16bf b, v8f c) {
  // (neg_a, A, neg_b, B, c_mod, C, reuse_a, reuse_b)
  return __builtin_amdgcn_wmma_f32_16x16x32_bf16(false, a, false, b, (short)0, c,
                                                 false, false);
}

// ---- gfx1250 async global->LDS copies (ASYNCcnt path), guarded ------------
#if defined(__has_builtin)
#if __has_builtin(__builtin_amdgcn_global_load_async_to_lds_b128) && \
    __has_builtin(__builtin_amdgcn_s_wait_asynccnt)
#define HAVE_ASYNC_LDS 1
#endif
#endif
#ifndef HAVE_ASYNC_LDS
#define HAVE_ASYNC_LDS 0
#endif

#if HAVE_ASYNC_LDS
__device__ __forceinline__ void async_copy16(const void* g, void* l) {
  // Builtin signature (from clang diagnostic): (v4i AS1*, v4i AS3*, Ii, Ii).
  // AS1 generic addresses are numerically identical; the AS3 offset is the
  // low 32 bits of the generic LDS address (aperture rule).
  __builtin_amdgcn_global_load_async_to_lds_b128(
      (__attribute__((address_space(1))) v4i*)(unsigned long long)(uintptr_t)g,
      (__attribute__((address_space(3))) v4i*)(unsigned int)(uintptr_t)l,
      0, 0);
}
__device__ __forceinline__ void async_wait0() {
  __builtin_amdgcn_s_wait_asynccnt(0);
}
#endif

// ---------------------------------------------------------------------------
// Generic tiled GEMM: C[M,N] = A[M,K] @ B[K,N]   (K % 32 == 0, M%128==0, N%128==0)
//   A_BF16: A is bf16 (else f32) ; C_BF16: store bf16 (else f32 + bias)
//   QSCALE: multiply columns < 512 by 0.125 (q-scale for the qkv projection)
// Block: 256 threads = 8 waves; tile 128x128x32; wave tile 32x64 (2x4 frags).
// ---------------------------------------------------------------------------
template<bool A_BF16, bool C_BF16, bool QSCALE>
__global__ __launch_bounds__(256) void gemm_wmma(const void* __restrict__ Ap,
                                                 const float* __restrict__ Bp,
                                                 const float* __restrict__ bias,
                                                 void* __restrict__ Cp,
                                                 int N, int K) {
  __shared__ __align__(16) unsigned short As[128][40];  // [m][k], pad->80B rows
  __shared__ __align__(16) unsigned short Bs[128][40];  // [n][k] (B transposed)

  const int tid  = threadIdx.x;
  const int lane = tid & 31;
  const int wid  = tid >> 5;
  const int wm   = wid & 3;     // 4 wave rows  (32 rows each)
  const int wn   = wid >> 2;    // 2 wave cols  (64 cols each)
  const int l15  = lane & 15;
  const int hi   = lane >> 4;

  const int m0 = blockIdx.y * 128;
  const int n0 = blockIdx.x * 128;

  v8f acc[2][4] = {};

  for (int kb = 0; kb < K; kb += 32) {
    __syncthreads();
    // ---- stage A tile (128x32) as bf16, 8 elements per slot ----
    for (int s = tid; s < 512; s += 256) {
      int row = s >> 2, kg = (s & 3) * 8;
      if (A_BF16) {
        const unsigned short* p =
            (const unsigned short*)Ap + (size_t)(m0 + row) * K + kb + kg;
#if HAVE_ASYNC_LDS
        async_copy16(p, &As[row][kg]);           // ASYNCcnt copy engine path
#else
        *(uint4*)&As[row][kg] = *(const uint4*)p;
#endif
      } else {
        const float* p = (const float*)Ap + (size_t)(m0 + row) * K + kb + kg;
        float4 f0 = *(const float4*)p;
        float4 f1 = *(const float4*)(p + 4);
        union { uint4 u; unsigned short h8[8]; } t;
        t.h8[0] = f2bf(f0.x); t.h8[1] = f2bf(f0.y);
        t.h8[2] = f2bf(f0.z); t.h8[3] = f2bf(f0.w);
        t.h8[4] = f2bf(f1.x); t.h8[5] = f2bf(f1.y);
        t.h8[6] = f2bf(f1.z); t.h8[7] = f2bf(f1.w);
        *(uint4*)&As[row][kg] = t.u;
      }
    }
    // ---- stage B tile (32x128) transposed to [n][k] bf16 ----
    for (int s = tid; s < 1024; s += 256) {
      int k = s >> 5, ng = (s & 31) * 4;
      const float* p = Bp + (size_t)(kb + k) * N + n0 + ng;
      float4 f = *(const float4*)p;
      Bs[ng + 0][k] = f2bf(f.x);
      Bs[ng + 1][k] = f2bf(f.y);
      Bs[ng + 2][k] = f2bf(f.z);
      Bs[ng + 3][k] = f2bf(f.w);
    }
#if HAVE_ASYNC_LDS
    if (A_BF16) async_wait0();                   // barrier does not flush async
#endif
    __syncthreads();

    if (kb + 32 < K) {  // hint next A tile into cache (global_prefetch path)
      const char* pf = (const char*)Ap +
          ((size_t)(m0 + (tid >> 1)) * K + kb + 32) * (A_BF16 ? 2 : 4);
      __builtin_prefetch(pf, 0, 1);
    }

    // ---- fragments ----
    BFrag af[2];
#pragma unroll
    for (int i = 0; i < 2; ++i) {
      int r = wm * 32 + i * 16 + l15;
      int ko = hi * 8;                           // interleaved A halves
      af[i].q[0] = *(const uint4*)&As[r][ko];
      af[i].q[1] = *(const uint4*)&As[r][ko + 16];
    }
    BFrag bf[4];
#pragma unroll
    for (int j = 0; j < 4; ++j) {
      int c = wn * 64 + j * 16 + l15;
      int ko = hi * 8;                           // interleaved B halves
      bf[j].q[0] = *(const uint4*)&Bs[c][ko];
      bf[j].q[1] = *(const uint4*)&Bs[c][ko + 16];
    }
#pragma unroll
    for (int i = 0; i < 2; ++i)
#pragma unroll
      for (int j = 0; j < 4; ++j)
        acc[i][j] = wmma_bf16(af[i].v, bf[j].v, acc[i][j]);
  }

  // ---- epilogue ----
#pragma unroll
  for (int i = 0; i < 2; ++i)
#pragma unroll
    for (int j = 0; j < 4; ++j)
#pragma unroll
      for (int r = 0; r < 8; ++r) {
        int grow = m0 + wm * 32 + i * 16 + r + hi * 8;
        int gcol = n0 + wn * 64 + j * 16 + l15;
        float v = acc[i][j][r];
        if (QSCALE && gcol < 512) v *= 0.125f;   // D^-0.5, D=64
        if (C_BF16) {
          ((unsigned short*)Cp)[(size_t)grow * N + gcol] = f2bf(v);
        } else {
          ((float*)Cp)[(size_t)grow * N + gcol] = v + bias[gcol];
        }
      }
}

// ---------------------------------------------------------------------------
// Attention kernel: one workgroup per (b,h,f).  n=196 padded to NP=224.
// LDS: KS[224][72], VT[64][232], QS[224][72], PW[8][16][232]  (bf16) = 150 KB
// ---------------------------------------------------------------------------
#define NTOK 196
#define NPAD 224
#define QKS  72      // QS/KS row stride (bf16 elems), 144B rows
#define PVS  232     // VT/PW row stride, 464B rows
#define SMEM_BYTES (size_t)((2 * 224 * QKS + 64 * PVS + 8 * 16 * PVS) * 2)

__global__ __launch_bounds__(256) void attn_wmma(
    const unsigned short* __restrict__ qkv, unsigned short* __restrict__ attn) {
  extern __shared__ __align__(16) char smem[];
  unsigned short* KS = (unsigned short*)smem;     // [224][72]
  unsigned short* VT = KS + 224 * QKS;            // [64][232]  (V transposed)
  unsigned short* QS = VT + 64 * PVS;             // [224][72]
  unsigned short* PW = QS + 224 * QKS;            // [8][16][232]

  const int tid  = threadIdx.x;
  const int lane = tid & 31;
  const int wid  = tid >> 5;
  const int l15  = lane & 15;
  const int hi   = lane >> 4;

  const int bhf = blockIdx.x;      // ((b*8)+h)*8 + f
  const int bb  = bhf >> 6;
  const int hh  = (bhf >> 3) & 7;
  const int ff  = bhf & 7;
  const int t0  = ff * NTOK;

  const size_t rowbase = (size_t)(bb * 1568 + t0) * 1536 + hh * 64;

  // ---- stage q,k,v tiles (already bf16, q pre-scaled) ----
#if HAVE_ASYNC_LDS
  // Q/K rows: straight 16B copies -> async copy engine (ASYNCcnt).
  for (int s = tid; s < NTOK * 8; s += 256) {
    int j = s >> 3, dg = (s & 7) * 8;
    const unsigned short* g = qkv + rowbase + (size_t)j * 1536 + dg;
    async_copy16(g,       &QS[j * QKS + dg]);
    async_copy16(g + 512, &KS[j * QKS + dg]);
  }
  {  // zero-fill padded rows 196..223
    uint4 z; z.x = z.y = z.z = z.w = 0u;
    for (int s = tid; s < (NPAD - NTOK) * 8; s += 256) {
      int j = NTOK + (s >> 3), dg = (s & 7) * 8;
      *(uint4*)&QS[j * QKS + dg] = z;
      *(uint4*)&KS[j * QKS + dg] = z;
    }
    // V needs a transpose -> stays on the VALU/DS path, overlaps the async Q/K
    for (int s = tid; s < NPAD * 8; s += 256) {
      int j = s >> 3, dg = (s & 7) * 8;
      uint4 vv = z;
      if (j < NTOK)
        vv = *(const uint4*)(qkv + rowbase + (size_t)j * 1536 + dg + 1024);
      union { uint4 u; unsigned short h8[8]; } t; t.u = vv;
#pragma unroll
      for (int e = 0; e < 8; ++e) VT[(dg + e) * PVS + j] = t.h8[e];
    }
  }
  async_wait0();                                  // flush ASYNCcnt before publish
  __syncthreads();
#else
  for (int s = tid; s < NPAD * 8; s += 256) {
    int j = s >> 3, dg = (s & 7) * 8;
    uint4 qv, kv, vv;
    if (j < NTOK) {
      const unsigned short* base = qkv + rowbase + (size_t)j * 1536 + dg;
      qv = *(const uint4*)(base);
      kv = *(const uint4*)(base + 512);
      vv = *(const uint4*)(base + 1024);
    } else {
      qv.x = qv.y = qv.z = qv.w = 0u; kv = qv; vv = qv;
    }
    *(uint4*)&QS[j * QKS + dg] = qv;
    *(uint4*)&KS[j * QKS + dg] = kv;
    union { uint4 u; unsigned short h8[8]; } t; t.u = vv;
#pragma unroll
    for (int e = 0; e < 8; ++e) VT[(dg + e) * PVS + j] = t.h8[e];
  }
  __syncthreads();
#endif

  unsigned short* pw = PW + wid * 16 * PVS;

  for (int rt = wid; rt < 13; rt += 8) {          // 13 row tiles cover 196 rows
    // A fragments of q rows (K=64 -> 2 steps)
    BFrag aq[2];
#pragma unroll
    for (int ks = 0; ks < 2; ++ks) {
      int r = rt * 16 + l15, ko = ks * 32 + hi * 8;
      aq[ks].q[0] = *(const uint4*)&QS[r * QKS + ko];
      aq[ks].q[1] = *(const uint4*)&QS[r * QKS + ko + 16];
    }
    // S = q k^T  (16 x 224)
    v8f s[14] = {};
#pragma unroll
    for (int ct = 0; ct < 14; ++ct)
#pragma unroll
      for (int ks = 0; ks < 2; ++ks) {
        BFrag bk;
        int c = ct * 16 + l15, ko = ks * 32 + hi * 8;
        bk.q[0] = *(const uint4*)&KS[c * QKS + ko];
        bk.q[1] = *(const uint4*)&KS[c * QKS + ko + 16];
        s[ct] = wmma_bf16(aq[ks].v, bk.v, s[ct]);
      }
    // mask padded key columns
#pragma unroll
    for (int ct = 12; ct < 14; ++ct)
      if (ct * 16 + l15 >= NTOK)
#pragma unroll
        for (int r = 0; r < 8; ++r) s[ct][r] = -3.0e38f;
    // rowwise softmax; each row lives in one 16-lane half
    float inv[8];
#pragma unroll
    for (int r = 0; r < 8; ++r) {
      float mx = s[0][r];
#pragma unroll
      for (int ct = 1; ct < 14; ++ct) mx = fmaxf(mx, s[ct][r]);
      for (int o = 1; o < 16; o <<= 1) mx = fmaxf(mx, __shfl_xor(mx, o, 32));
      float sum = 0.f;
#pragma unroll
      for (int ct = 0; ct < 14; ++ct) {
        float e = __expf(s[ct][r] - mx);
        s[ct][r] = e;
        sum += e;
      }
      for (int o = 1; o < 16; o <<= 1) sum += __shfl_xor(sum, o, 32);
      inv[r] = 1.0f / sum;
    }
    // unnormalized P -> per-wave LDS pane (bf16)
#pragma unroll
    for (int ct = 0; ct < 14; ++ct)
#pragma unroll
      for (int r = 0; r < 8; ++r)
        pw[(r + hi * 8) * PVS + ct * 16 + l15] = f2bf(s[ct][r]);

    // O = P @ V   (16 x 64), 1/rowsum folded into the store
#pragma unroll
    for (int dt = 0; dt < 4; ++dt) {
      v8f o = {};
#pragma unroll
      for (int ks = 0; ks < 7; ++ks) {
        BFrag ap, bv;
        int ko = ks * 32;
        ap.q[0] = *(const uint4*)&pw[l15 * PVS + ko + hi * 8];
        ap.q[1] = *(const uint4*)&pw[l15 * PVS + ko + hi * 8 + 16];
        int d = dt * 16 + l15;
        bv.q[0] = *(const uint4*)&VT[d * PVS + ko + hi * 8];
        bv.q[1] = *(const uint4*)&VT[d * PVS + ko + hi * 8 + 16];
        o = wmma_bf16(ap.v, bv.v, o);
      }
#pragma unroll
      for (int r = 0; r < 8; ++r) {
        int tok = rt * 16 + r + hi * 8;
        if (tok < NTOK) {
          size_t off = ((size_t)(bb * 1568 + t0 + tok)) * 512 +
                       hh * 64 + dt * 16 + l15;
          attn[off] = f2bf(o[r] * inv[r]);
        }
      }
    }
  }
}

// ---------------------------------------------------------------------------
extern "C" void kernel_launch(void* const* d_in, const int* in_sizes, int n_in,
                              void* d_out, int out_size, void* d_ws, size_t ws_size,
                              hipStream_t stream) {
  (void)in_sizes; (void)n_in; (void)out_size; (void)ws_size;
  const float* x    = (const float*)d_in[0];  // [8,1568,512]
  const float* Wqkv = (const float*)d_in[1];  // [512,1536]
  const float* Wout = (const float*)d_in[2];  // [512,512]
  const float* bout = (const float*)d_in[3];  // [512]
  float* out = (float*)d_out;                 // [8,1568,512]

  unsigned short* qkv  = (unsigned short*)d_ws;            // bf16 [12544,1536]
  unsigned short* attn = qkv + (size_t)12544 * 1536;       // bf16 [12544,512]

  (void)hipFuncSetAttribute((const void*)attn_wmma,
                            hipFuncAttributeMaxDynamicSharedMemorySize,
                            (int)SMEM_BYTES);

  // k1: qkv projection, M=12544 N=1536 K=512, q columns scaled by 1/8
  gemm_wmma<false, true, true>
      <<<dim3(12, 98), 256, 0, stream>>>(x, Wqkv, nullptr, qkv, 1536, 512);
  // k2: axial attention per (b,h,f)
  attn_wmma<<<dim3(512), 256, SMEM_BYTES, stream>>>(qkv, attn);
  // k3: output projection + bias, M=12544 N=512 K=512
  gemm_wmma<true, false, false>
      <<<dim3(4, 98), 256, 0, stream>>>(attn, Wout, bout, out, 512, 512);
}
